// MLA_82583631167701
// MI455X (gfx1250) — matrix operand order repaired and checked by reading, not compiled, not run on hardware
//
#include <hip/hip_runtime.h>
#include <hip/hip_bf16.h>

// ---------------------------------------------------------------------------
// MLA forward for MI455X (gfx1250, wave32).
// - All contractions on V_WMMA_F32_16X16X32_F16 (f16 in, fp32 accumulate).
// - Projection GEMMs write q/k as f16 row-major and v as f16 transposed
//   (b,h,d,s), so the attention kernel streams K/V/Q tiles into LDS with the
//   Tensor Data Mover (tensor_load_to_lds, TENSORcnt), double-buffered to
//   overlap DMA with WMMA.
// Pipeline:
//   1. qh = f16( x  @ wq^T )          4096x1024x1024
//   2. kl =      x  @ wk_a^T          4096x256x1024   (fp32)
//   3. vl =      x  @ wv_a^T
//   4. kh = f16( kl @ wk_b^T )        4096x1024x256
//   5. vt = f16t(vl @ wv_b^T )        (b,h,d,s) layout
//   6. head-indexed RoPE (reference quirk) on dims [0,16) of qh, kh
//   7. flash attention per (b,h): softmax(q k^T / 8) v  -> ao (fp32)
//   8. out = ao @ wo^T
// Workspace: 3*M*DIM f16 + 2*M*LAT fp32 + M*DIM fp32  = ~48 MB.
// ---------------------------------------------------------------------------

#define BATCH 2
#define SEQ   2048
#define NH    16
#define HD    64
#define DIM   1024
#define LAT   256

typedef __attribute__((ext_vector_type(16))) _Float16     v16h;
typedef __attribute__((ext_vector_type(8)))  _Float16     v8h;
typedef __attribute__((ext_vector_type(8)))  float        v8f;
typedef __attribute__((ext_vector_type(4)))  unsigned int v4u;
typedef __attribute__((ext_vector_type(4)))  int          v4i;
typedef __attribute__((ext_vector_type(8)))  int          v8i;

#define WMMA_F16(a, b, c) \
  __builtin_amdgcn_wmma_f32_16x16x32_f16(false, (a), false, (b), (short)0, (c), false, false)

// Pack two float4 into 8 halves.
__device__ __forceinline__ v8h cvt2(float4 a, float4 b) {
  v8h r;
  r[0] = (_Float16)a.x; r[1] = (_Float16)a.y; r[2] = (_Float16)a.z; r[3] = (_Float16)a.w;
  r[4] = (_Float16)b.x; r[5] = (_Float16)b.y; r[6] = (_Float16)b.z; r[7] = (_Float16)b.w;
  return r;
}

// A-fragment (16x32 f16, M on lanes): element e -> k = koff + 16*(e>=8) + 8*(lane>=16) + (e&7)
__device__ __forceinline__ v16h load_frag_a(const _Float16* base, int stride, int row0, int koff) {
  const int lane = threadIdx.x & 31;
  const int m  = row0 + (lane & 15);
  const int hi = lane >> 4;
  const _Float16* p = base + (size_t)m * stride + koff + hi * 8;
  union { v16h v; v8h h[2]; } u;
  u.h[0] = *(const v8h*)(p);
  u.h[1] = *(const v8h*)(p + 16);
  return u.v;
}

// B-fragment (32x16 f16, N on lanes): element e -> k = koff + 16*(e>=8) + 8*(lane>=16) + (e&7)
__device__ __forceinline__ v16h load_frag_b(const _Float16* base, int stride, int col0, int koff) {
  const int lane = threadIdx.x & 31;
  const int n  = col0 + (lane & 15);
  const int hi = lane >> 4;
  const _Float16* p = base + (size_t)n * stride + koff + hi * 8;
  union { v16h v; v8h h[2]; } u;
  u.h[0] = *(const v8h*)(p);
  u.h[1] = *(const v8h*)(p + 16);
  return u.v;
}

// ---------------------------------------------------------------------------
// Tensor Data Mover: 2D tile (f16 elements) global -> LDS, with per-row LDS
// padding so the LDS image matches our padded fragment strides.
// Descriptor bitfields per CDNA5 ISA ch.8.3/8.4:
//   group0: [1:0]=count(1), [63:32]=lds_addr, [120:64]=global byte addr,
//           [127:126]=type(2)
//   group1: [17:16]=data_size(1 => 2B), [20]=pad_enable,
//           [24:22]=pad_interval (0=2 dw,1=4,...  4=32dw=128B, 3=16dw=64B),
//           [31:25]=pad_amount (3 => 4 dwords = 16B),
//           [79:48]=tensor_dim0, [111:80]=tensor_dim1 (== tile dims, no OOB),
//           [127:112]=tile_dim0, [143:128]=tile_dim1, [159:144]=tile_dim2(0),
//           [207:160]=tensor_dim0_stride (elements)
// Remaining groups zero (2D tile; higher dims / gather unused).
// This toolchain exposes the 6-arg builtin:
//   (uint32x4 g0, int32x8 g1, int32x4, int32x4, int32x8, i32 cpol)
// ---------------------------------------------------------------------------
__device__ __forceinline__ void tdm_load_2d(unsigned lds_addr, const _Float16* gptr,
                                            unsigned tile0, unsigned tile1,
                                            unsigned stride_elems,
                                            unsigned pad_interval_code,
                                            unsigned pad_amount_code) {
  const unsigned long long ga = (unsigned long long)(uintptr_t)gptr;
  v4u g0;
  g0[0] = 1u;                                     // count = 1 (valid descriptor)
  g0[1] = lds_addr;                               // LDS byte address
  g0[2] = (unsigned)(ga & 0xffffffffull);         // global addr [95:64]
  g0[3] = (unsigned)(ga >> 32) | 0x80000000u;     // global addr hi | type=2
  v8i g1;
  g1[0] = (int)((1u << 16) | (1u << 20) |
                (pad_interval_code << 22) | (pad_amount_code << 25));
  g1[1] = (int)((tile0 & 0xffffu) << 16);         // tensor_dim0 lo16 @ [63:48]
  g1[2] = (int)((tile1 & 0xffffu) << 16);         // tensor_dim1 lo16 @ [95:80]
  g1[3] = (int)((tile0 & 0xffffu) << 16);         // tile_dim0      @ [127:112]
  g1[4] = (int)(tile1 & 0xffffu);                 // tile_dim1; tile_dim2 = 0
  g1[5] = (int)stride_elems;                      // tensor_dim0_stride lo32
  g1[6] = 0;
  g1[7] = 0;
  const v4i z4 = {};
  const v8i z8 = {};
  __builtin_amdgcn_tensor_load_to_lds(g0, g1, z4, z4, z8, 0);
}

// ---------------------------------------------------------------------------
// C[M,N] = A[M,K] @ W[N,K]^T.  fp32 inputs, f16 WMMA, fp32 accumulate.
// OMODE 0: fp32 row-major out; 1: f16 row-major out; 2: f16 out in
// (b,h,d,s) "transposed V" layout.
// ---------------------------------------------------------------------------
template <int OMODE>
__global__ __launch_bounds__(128) void gemm_nt_f16(
    const float* __restrict__ A, const float* __restrict__ W,
    void* __restrict__ Cout, int M, int N, int K)
{
  __shared__ alignas(16) _Float16 sA[64][40];
  __shared__ alignas(16) _Float16 sB[64][40];

  const int t    = threadIdx.x;
  const int wave = t >> 5;
  const int lane = t & 31;
  const int nl   = lane & 15;
  const int hi   = lane >> 4;
  const int wm   = (wave >> 1) * 32;
  const int wn   = (wave & 1) * 32;
  const int mblk = blockIdx.y * 64;
  const int nblk = blockIdx.x * 64;
  const int lrow = t >> 1;
  const int lkk  = (t & 1) * 16;

  v8f acc00 = {}, acc01 = {}, acc10 = {}, acc11 = {};

  for (int k0 = 0; k0 < K; k0 += 32) {
    const float4* ga = reinterpret_cast<const float4*>(A + (size_t)(mblk + lrow) * K + k0 + lkk);
    const float4* gw = reinterpret_cast<const float4*>(W + (size_t)(nblk + lrow) * K + k0 + lkk);
    float4 fa0 = ga[0], fa1 = ga[1], fa2 = ga[2], fa3 = ga[3];
    float4 fw0 = gw[0], fw1 = gw[1], fw2 = gw[2], fw3 = gw[3];
    if (k0 + 32 < K) {
      __builtin_prefetch(A + (size_t)(mblk + lrow) * K + k0 + 32 + lkk, 0, 1);
      __builtin_prefetch(W + (size_t)(nblk + lrow) * K + k0 + 32 + lkk, 0, 1);
    }
    *(v8h*)&sA[lrow][lkk]     = cvt2(fa0, fa1);
    *(v8h*)&sA[lrow][lkk + 8] = cvt2(fa2, fa3);
    *(v8h*)&sB[lrow][lkk]     = cvt2(fw0, fw1);
    *(v8h*)&sB[lrow][lkk + 8] = cvt2(fw2, fw3);
    __syncthreads();

    v16h a0 = load_frag_a(&sA[0][0], 40, wm,      0);
    v16h a1 = load_frag_a(&sA[0][0], 40, wm + 16, 0);
    v16h b0 = load_frag_b(&sB[0][0], 40, wn,      0);
    v16h b1 = load_frag_b(&sB[0][0], 40, wn + 16, 0);
    acc00 = WMMA_F16(a0, b0, acc00);
    acc01 = WMMA_F16(a0, b1, acc01);
    acc10 = WMMA_F16(a1, b0, acc10);
    acc11 = WMMA_F16(a1, b1, acc11);
    __syncthreads();
  }

  #pragma unroll
  for (int r = 0; r < 8; ++r) {
    const int gm = mblk + wm + r + hi * 8;
    const int gn = nblk + wn + nl;
    if (OMODE == 0) {
      float* C = (float*)Cout;
      C[(size_t)gm * N + gn]             = acc00[r];
      C[(size_t)gm * N + gn + 16]        = acc01[r];
      C[(size_t)(gm + 16) * N + gn]      = acc10[r];
      C[(size_t)(gm + 16) * N + gn + 16] = acc11[r];
    } else if (OMODE == 1) {
      _Float16* C = (_Float16*)Cout;
      C[(size_t)gm * N + gn]             = (_Float16)acc00[r];
      C[(size_t)gm * N + gn + 16]        = (_Float16)acc01[r];
      C[(size_t)(gm + 16) * N + gn]      = (_Float16)acc10[r];
      C[(size_t)(gm + 16) * N + gn + 16] = (_Float16)acc11[r];
    } else {
      // vt[((b*NH+h)*HD + d) * SEQ + s]; row gm = b*SEQ + s, col gn = h*HD + d.
      _Float16* C = (_Float16*)Cout;
      const size_t s0 = (size_t)(gm & (SEQ - 1)) + (size_t)(gm >> 11) * DIM * SEQ;
      const size_t s1 = (size_t)((gm + 16) & (SEQ - 1)) + (size_t)((gm + 16) >> 11) * DIM * SEQ;
      C[(size_t)gn * SEQ + s0]        = (_Float16)acc00[r];
      C[(size_t)(gn + 16) * SEQ + s0] = (_Float16)acc01[r];
      C[(size_t)gn * SEQ + s1]        = (_Float16)acc10[r];
      C[(size_t)(gn + 16) * SEQ + s1] = (_Float16)acc11[r];
    }
  }
}

// ---------------------------------------------------------------------------
// Reference RoPE quirk: angle depends on the HEAD index (freq table indexed
// along the head axis after the permute). Rotates dims [0,16) in place (f16).
// ---------------------------------------------------------------------------
__global__ void rope_kernel_f16(_Float16* __restrict__ q, _Float16* __restrict__ k)
{
  const int idx = blockIdx.x * blockDim.x + threadIdx.x;   // over B*S*NH
  if (idx >= BATCH * SEQ * NH) return;
  const int h = idx % NH;
  const size_t base = (size_t)(idx / NH) * DIM + (size_t)h * HD;
  _Float16* pq = q + base;
  _Float16* pk = k + base;
  #pragma unroll
  for (int j = 0; j < 8; ++j) {
    const float freq = __expf(-(float)(2 * j) * (9.210340371976184f / 64.f)); // 10000^(-2j/64)
    const float ang = (float)h * freq;
    const float c = __cosf(ang), s = __sinf(ang);
    float a, b;
    a = (float)pq[2 * j]; b = (float)pq[2 * j + 1];
    pq[2 * j]     = (_Float16)(a * c - b * s);
    pq[2 * j + 1] = (_Float16)(a * s + b * c);
    a = (float)pk[2 * j]; b = (float)pk[2 * j + 1];
    pk[2 * j]     = (_Float16)(a * c - b * s);
    pk[2 * j + 1] = (_Float16)(a * s + b * c);
  }
}

// ---------------------------------------------------------------------------
// Flash attention, one block per (b, h, 64 q-rows); 4 waves x 16 q-rows.
// K/V tiles (32 keys) DMA'd into double-buffered LDS by the TDM; wave 0
// issues descriptors for tile i+1, waits TENSORcnt<=2 (in-order per wave ->
// tile i complete), so the DMA of tile i+1 overlaps the WMMAs of tile i.
// Dynamic-LDS byte layout (base offset 0, all 16B aligned):
//   sQ       @     0  : 64 rows x 72 halves  (9216 B)   row = 128B data + 16B pad
//   sK[0]    @  9216  : 32 x 72              (4608 B)
//   sK[1]    @ 13824
//   sVt[0]   @ 18432  : 64 dims x 40 halves  (5120 B)   row = 64B data + 16B pad
//   sVt[1]   @ 23552
//   sP       @ 28672  : 4 waves x 16 x 40    (5120 B)
//   total 33792 B
// ---------------------------------------------------------------------------
#define LQ_OFF  0u
#define LK_OFF  9216u
#define LK_SZ   4608u
#define LV_OFF  18432u
#define LV_SZ   5120u
#define LP_OFF  28672u
#define ATT_LDS 33792u

__global__ __launch_bounds__(128) void mla_attn(
    const _Float16* __restrict__ Qh, const _Float16* __restrict__ Kh,
    const _Float16* __restrict__ Vt, float* __restrict__ O)
{
  extern __shared__ char smem[];
  const int t    = threadIdx.x;
  const int wave = t >> 5;
  const int lane = t & 31;
  const int nl   = lane & 15;
  const int hi   = lane >> 4;
  const int qblk = blockIdx.x * 64;
  const int h    = blockIdx.y;
  const int b    = blockIdx.z;
  const size_t bofs  = (size_t)b * SEQ * DIM + (size_t)h * HD;       // (b,s,h*64+d)
  const size_t vbofs = (size_t)(b * NH + h) * HD * SEQ;              // (b,h,d,s)

  // Prologue: DMA the q-block (64x64) and key tile 0 (K 32x64, Vt 64x32).
  if (wave == 0) {
    tdm_load_2d(LQ_OFF, Qh + bofs + (size_t)qblk * DIM, 64, 64, DIM, /*128B rows*/4, 3);
    tdm_load_2d(LK_OFF, Kh + bofs, 64, 32, DIM, 4, 3);
    tdm_load_2d(LV_OFF, Vt + vbofs, 32, 64, SEQ, /*64B rows*/3, 3);
    __builtin_amdgcn_s_wait_tensorcnt(0);
  }
  __syncthreads();

  const _Float16* sQ = (const _Float16*)(smem + LQ_OFF);
  const v16h qa0 = load_frag_a(sQ, 72, wave * 16, 0);    // dims  0..31
  const v16h qa1 = load_frag_a(sQ, 72, wave * 16, 32);   // dims 32..63

  v8f oacc[4] = {};
  float rm[8], rl[8];
  #pragma unroll
  for (int r = 0; r < 8; ++r) { rm[r] = -1e30f; rl[r] = 0.f; }

  for (int it = 0; it < SEQ / 32; ++it) {
    const int cur = it & 1;
    if (wave == 0) {
      if (it + 1 < SEQ / 32) {   // prefetch next tile into the other buffer
        const int kt = (it + 1) * 32;
        tdm_load_2d(LK_OFF + (cur ^ 1) * LK_SZ, Kh + bofs + (size_t)kt * DIM, 64, 32, DIM, 4, 3);
        tdm_load_2d(LV_OFF + (cur ^ 1) * LV_SZ, Vt + vbofs + kt, 32, 64, SEQ, 3, 3);
        __builtin_amdgcn_s_wait_tensorcnt(2);   // in-order: current tile landed
      } else {
        __builtin_amdgcn_s_wait_tensorcnt(0);
      }
    }
    __syncthreads();

    const _Float16* sK  = (const _Float16*)(smem + LK_OFF + cur * LK_SZ);
    const _Float16* sVt = (const _Float16*)(smem + LV_OFF + cur * LV_SZ);
    _Float16*       sPw = (_Float16*)(smem + LP_OFF) + wave * (16 * 40);

    // S = Q K^T for 16 q-rows x 32 keys.
    const v16h kb00 = load_frag_b(sK, 72, 0,  0);
    const v16h kb01 = load_frag_b(sK, 72, 0,  32);
    const v16h kb10 = load_frag_b(sK, 72, 16, 0);
    const v16h kb11 = load_frag_b(sK, 72, 16, 32);
    v8f s0 = {}, s1 = {};
    s0 = WMMA_F16(qa0, kb00, s0);
    s0 = WMMA_F16(qa1, kb01, s0);
    s1 = WMMA_F16(qa0, kb10, s1);
    s1 = WMMA_F16(qa1, kb11, s1);

    // Online softmax: row m = r + 8*hi lives on the 16 lanes of one half-wave.
    float alpha[8];
    #pragma unroll
    for (int r = 0; r < 8; ++r) {
      const float e0 = s0[r] * 0.125f;            // 1/sqrt(64)
      const float e1 = s1[r] * 0.125f;
      float m = fmaxf(e0, e1);
      m = fmaxf(m, __shfl_xor(m, 1));
      m = fmaxf(m, __shfl_xor(m, 2));
      m = fmaxf(m, __shfl_xor(m, 4));
      m = fmaxf(m, __shfl_xor(m, 8));
      const float mn = fmaxf(rm[r], m);
      alpha[r] = __expf(rm[r] - mn);
      rm[r] = mn;
      const float p0 = __expf(e0 - mn);
      const float p1 = __expf(e1 - mn);
      const int mrow = r + hi * 8;
      sPw[mrow * 40 + nl]      = (_Float16)p0;    // C-layout -> A-layout via LDS
      sPw[mrow * 40 + nl + 16] = (_Float16)p1;
      float rs = p0 + p1;
      rs += __shfl_xor(rs, 1);
      rs += __shfl_xor(rs, 2);
      rs += __shfl_xor(rs, 4);
      rs += __shfl_xor(rs, 8);
      rl[r] = rl[r] * alpha[r] + rs;
    }
    #pragma unroll
    for (int f = 0; f < 4; ++f)
      #pragma unroll
      for (int r = 0; r < 8; ++r)
        oacc[f][r] *= alpha[r];

    // O += P V  (contraction = 32 keys = one full f16 WMMA K).
    const v16h pa = load_frag_a(sPw, 40, 0, 0);
    #pragma unroll
    for (int f = 0; f < 4; ++f) {
      const v16h vb = load_frag_b(sVt, 40, f * 16, 0);
      oacc[f] = WMMA_F16(pa, vb, oacc[f]);
    }
    __syncthreads();   // all fragment reads done before buffer reuse
  }

  #pragma unroll
  for (int f = 0; f < 4; ++f) {
    #pragma unroll
    for (int r = 0; r < 8; ++r) {
      const int mrow = r + hi * 8;
      const size_t addr = bofs + (size_t)(qblk + wave * 16 + mrow) * DIM + f * 16 + nl;
      O[addr] = oacc[f][r] / rl[r];
    }
  }
}

// ---------------------------------------------------------------------------
extern "C" void kernel_launch(void* const* d_in, const int* in_sizes, int n_in,
                              void* d_out, int out_size, void* d_ws, size_t ws_size,
                              hipStream_t stream) {
  (void)in_sizes; (void)n_in; (void)out_size; (void)ws_size;
  const float* x    = (const float*)d_in[0];
  const float* wq   = (const float*)d_in[1];
  const float* wk_a = (const float*)d_in[2];
  const float* wk_b = (const float*)d_in[3];
  const float* wv_a = (const float*)d_in[4];
  const float* wv_b = (const float*)d_in[5];
  const float* wo   = (const float*)d_in[6];
  float* out = (float*)d_out;

  const int M = BATCH * SEQ;                               // 4096
  _Float16* qh = (_Float16*)d_ws;                          // M*DIM f16
  _Float16* kh = qh + (size_t)M * DIM;                     // M*DIM f16
  _Float16* vt = kh + (size_t)M * DIM;                     // M*DIM f16 (b,h,d,s)
  float*    kl = (float*)(vt + (size_t)M * DIM);           // M*LAT fp32
  float*    vl = kl + (size_t)M * LAT;                     // M*LAT fp32
  float*    ao = vl + (size_t)M * LAT;                     // M*DIM fp32

  const dim3 blk(128);
  gemm_nt_f16<1><<<dim3(DIM / 64, M / 64), blk, 0, stream>>>(x,  wq,   qh, M, DIM, DIM);
  gemm_nt_f16<0><<<dim3(LAT / 64, M / 64), blk, 0, stream>>>(x,  wk_a, kl, M, LAT, DIM);
  gemm_nt_f16<0><<<dim3(LAT / 64, M / 64), blk, 0, stream>>>(x,  wv_a, vl, M, LAT, DIM);
  gemm_nt_f16<1><<<dim3(DIM / 64, M / 64), blk, 0, stream>>>(kl, wk_b, kh, M, DIM, LAT);
  gemm_nt_f16<2><<<dim3(DIM / 64, M / 64), blk, 0, stream>>>(vl, wv_b, vt, M, DIM, LAT);
  rope_kernel_f16<<<(BATCH * SEQ * NH + 255) / 256, 256, 0, stream>>>(qh, kh);
  mla_attn<<<dim3(SEQ / 64, NH, BATCH), blk, ATT_LDS, stream>>>(qh, kh, vt, ao);
  gemm_nt_f16<0><<<dim3(DIM / 64, M / 64), blk, 0, stream>>>(ao, wo, out, M, DIM, DIM);
}
